// Critic_64364379898149
// MI455X (gfx1250) — matrix operand order repaired and checked by reading, not compiled, not run on hardware
//
#include <hip/hip_runtime.h>

typedef __attribute__((ext_vector_type(16))) _Float16 v16h;
typedef __attribute__((ext_vector_type(8)))  _Float16 v8h;
typedef __attribute__((ext_vector_type(8)))  float    v8f;

#define NPTS 10000
#define NBATCH 8

// ---------------- elementwise f32 -> f16 (used for level-0 feats: (B,n,3) point-major) ----
__global__ void f32_to_f16_kernel(const float* __restrict__ in, _Float16* __restrict__ out,
                                  size_t n) {
  size_t i = (size_t)blockIdx.x * blockDim.x + threadIdx.x;
  if (i < n) out[i] = (_Float16)in[i];
}

// ---------------- farthest point sampling: one block per batch ----------------
__global__ void fps_kernel(const float* __restrict__ xyz, float* __restrict__ mind,
                           float* __restrict__ cent, int n, int npoint) {
  int b = blockIdx.x;
  const float* X = xyz + (size_t)b * n * 3;
  float* md = mind + (size_t)b * n;
  float* C = cent + (size_t)b * npoint * 3;
  int t = threadIdx.x, nt = blockDim.x;
  __shared__ float s_val[512];
  __shared__ int s_idx[512];
  __shared__ int s_cur;
  for (int i = t; i < n; i += nt) md[i] = 3.4e38f;
  if (t == 0) s_cur = 0;
  __syncthreads();
  for (int it = 0; it < npoint; ++it) {
    int cur = s_cur;
    float cx = X[cur * 3 + 0], cy = X[cur * 3 + 1], cz = X[cur * 3 + 2];
    if (t == 0) { C[it * 3 + 0] = cx; C[it * 3 + 1] = cy; C[it * 3 + 2] = cz; }
    float best = -1.f; int bi = 0;
    for (int i = t; i < n; i += nt) {
      float dx = X[i * 3] - cx, dy = X[i * 3 + 1] - cy, dz = X[i * 3 + 2] - cz;
      float d = dx * dx + dy * dy + dz * dz;
      float m = md[i]; m = fminf(m, d); md[i] = m;
      if (m > best) { best = m; bi = i; }
    }
    s_val[t] = best; s_idx[t] = bi;
    __syncthreads();
    for (int s = nt >> 1; s > 0; s >>= 1) {
      if (t < s) {
        if (s_val[t + s] > s_val[t] ||
            (s_val[t + s] == s_val[t] && s_idx[t + s] < s_idx[t])) {
          s_val[t] = s_val[t + s]; s_idx[t] = s_idx[t + s];
        }
      }
      __syncthreads();
    }
    if (t == 0) s_cur = s_idx[0];
    __syncthreads();
  }
}

// ---------------- ball query: first nsample indices within radius ----------------
__global__ void ball_query_kernel(const float* __restrict__ xyz, const float* __restrict__ cent,
                                  int* __restrict__ bidx, int n, int npoint, float r2,
                                  int nsample) {
  int j = blockIdx.x * blockDim.x + threadIdx.x;
  int b = blockIdx.y;
  if (j >= npoint) return;
  const float* X = xyz + (size_t)b * n * 3;
  const float* c = cent + ((size_t)b * npoint + j) * 3;
  int* out = bidx + ((size_t)b * npoint + j) * nsample;
  float cx = c[0], cy = c[1], cz = c[2];
  int cnt = 0, first = 0;
  for (int i = 0; i < n && cnt < nsample; ++i) {
    float dx = X[i * 3] - cx, dy = X[i * 3 + 1] - cy, dz = X[i * 3 + 2] - cz;
    if (dx * dx + dy * dy + dz * dz < r2) { if (cnt == 0) first = i; out[cnt++] = i; }
  }
  for (int k = cnt; k < nsample; ++k) out[k] = first;
}

// -------- build grouped MLP input, column-major [col][kpad], zero-padded channels --------
__global__ void group_kernel(const float* __restrict__ xyz, const float* __restrict__ cent,
                             const int* __restrict__ bidx, const _Float16* __restrict__ feats,
                             _Float16* __restrict__ Xo, int n_src, int npoint, int nsample,
                             int cf, int kpad, size_t fstride, size_t xstride) {
  int col = blockIdx.x * blockDim.x + threadIdx.x;
  int b = blockIdx.y;
  int ncols = npoint * nsample;
  if (col >= ncols) return;
  int j = col / nsample;
  int i = bidx[((size_t)b * npoint + j) * nsample + (col % nsample)];
  const float* X = xyz + (size_t)b * n_src * 3;
  const float* c = cent + ((size_t)b * npoint + j) * 3;
  _Float16* o = Xo + (size_t)b * xstride + (size_t)col * kpad;
  o[0] = (_Float16)(X[i * 3 + 0] - c[0]);
  o[1] = (_Float16)(X[i * 3 + 1] - c[1]);
  o[2] = (_Float16)(X[i * 3 + 2] - c[2]);
  const _Float16* f = feats + (size_t)b * fstride + (size_t)i * cf;
  for (int cc = 0; cc < cf; ++cc) o[3 + cc] = f[cc];
  for (int cc = 3 + cf; cc < kpad; ++cc) o[cc] = (_Float16)0.f;
}

// ---------------- weight prep: f32 [cout][cin] -> f16 [cout][kpad], zero-padded ----------
__global__ void prep_w_kernel(const float* __restrict__ W, _Float16* __restrict__ Wh,
                              int cout, int cin, int kpad) {
  int idx = blockIdx.x * blockDim.x + threadIdx.x;
  if (idx >= cout * kpad) return;
  int r = idx / kpad, k = idx % kpad;
  Wh[idx] = (k < cin) ? (_Float16)W[(size_t)r * cin + k] : (_Float16)0.f;
}

// ---------------- WMMA GEMM: Y[col][cout] = relu((Wh @ X[col]) * s + b) -----------------
// Wh: f16 [cout][kdim] (padded); X: f16 col-major [ncols][kdim]; cout%16==0, ncols%16==0,
// kdim%32==0. block = 128 threads = 4 waves, one 16x16 D tile per wave.
__global__ void gemm_wmma_kernel(const _Float16* __restrict__ Wh, const float* __restrict__ S,
                                 const float* __restrict__ Bv, const _Float16* __restrict__ Xin,
                                 _Float16* __restrict__ Yout, int cout, int kdim, int ncols,
                                 size_t xstride, size_t ystride) {
  int lane = threadIdx.x & 31;
  int wv = threadIdx.x >> 5;
  int m0 = (blockIdx.x * 4 + wv) * 16;
  int n0 = blockIdx.y * 16;
  int b = blockIdx.z;
  if (m0 >= cout) return;  // wave-uniform
  int row = m0 + (lane & 15);
  int akb = (lane >> 4) * 8;   // A K-base within 32-chunk for this half-wave
  int col = n0 + (lane & 15);
  int bkb = (lane >> 4) * 8;   // B K-base within 32-chunk for this half-wave
  const _Float16* ap = Wh + (size_t)row * kdim;
  const _Float16* bp = Xin + (size_t)b * xstride + (size_t)col * kdim;
  v8f acc = {};
  for (int k0 = 0; k0 < kdim; k0 += 32) {
    v8h al = *(const v8h*)(ap + k0 + akb);
    v8h ah = *(const v8h*)(ap + k0 + akb + 16);
    v8h bl = *(const v8h*)(bp + k0 + bkb);
    v8h bh = *(const v8h*)(bp + k0 + bkb + 16);
    v16h a = __builtin_shufflevector(al, ah, 0, 1, 2, 3, 4, 5, 6, 7, 8, 9, 10, 11, 12, 13, 14, 15);
    v16h bb = __builtin_shufflevector(bl, bh, 0, 1, 2, 3, 4, 5, 6, 7, 8, 9, 10, 11, 12, 13, 14, 15);
    acc = __builtin_amdgcn_wmma_f32_16x16x32_f16(false, a, false, bb, (short)0, acc,
                                                 false, false);
  }
  int mbase = m0 + (lane >> 4) * 8;
  v8h yo;
#pragma unroll
  for (int r = 0; r < 8; ++r) {
    float v = acc[r] * S[mbase + r] + Bv[mbase + r];
    yo[r] = (_Float16)fmaxf(v, 0.f);
  }
  *(v8h*)(Yout + (size_t)b * ystride + (size_t)col * cout + mbase) = yo;
}

// ---------------- max-pool over nsample columns (col-major in/out) ----------------
__global__ void maxpool_kernel(const _Float16* __restrict__ Xin, _Float16* __restrict__ Fout,
                               int cout, int npoint, int nsample, size_t xstride,
                               size_t fstride) {
  int idx = blockIdx.x * blockDim.x + threadIdx.x;
  int b = blockIdx.y;
  if (idx >= cout * npoint) return;
  int p = idx / cout, co = idx % cout;
  const _Float16* X = Xin + (size_t)b * xstride + (size_t)p * nsample * cout + co;
  float m = -3.4e38f;
  for (int k = 0; k < nsample; ++k) m = fmaxf(m, (float)X[(size_t)k * cout]);
  Fout[(size_t)b * fstride + (size_t)p * cout + co] = (_Float16)m;
}

// ------- FP: 3-NN interpolate + concat skip -> X col-major [j][c2+c1] (c1+c2 %32==0) -----
__global__ void fp_interp_kernel(const float* __restrict__ unk, const float* __restrict__ knw,
                                 const _Float16* __restrict__ skip,
                                 const _Float16* __restrict__ kfeat,
                                 _Float16* __restrict__ Xo, int n, int m, int c1, int c2,
                                 size_t sstride, size_t kstride, size_t xstride) {
  int j = blockIdx.x * blockDim.x + threadIdx.x;
  int b = blockIdx.y;
  if (j >= n) return;
  const float* U = unk + ((size_t)b * n + j) * 3;
  const float* K = knw + (size_t)b * m * 3;
  float ux = U[0], uy = U[1], uz = U[2];
  float d0 = 3.4e38f, d1 = 3.4e38f, d2 = 3.4e38f;
  int i0 = 0, i1 = 0, i2 = 0;
  for (int i = 0; i < m; ++i) {
    float dx = K[i * 3] - ux, dy = K[i * 3 + 1] - uy, dz = K[i * 3 + 2] - uz;
    float d = dx * dx + dy * dy + dz * dz;
    if (d < d0)      { d2 = d1; i2 = i1; d1 = d0; i1 = i0; d0 = d; i0 = i; }
    else if (d < d1) { d2 = d1; i2 = i1; d1 = d; i1 = i; }
    else if (d < d2) { d2 = d; i2 = i; }
  }
  float w0 = 1.f / (d0 + 1e-8f), w1 = 1.f / (d1 + 1e-8f), w2 = 1.f / (d2 + 1e-8f);
  float ws = w0 + w1 + w2; w0 /= ws; w1 /= ws; w2 /= ws;
  const _Float16* K0 = kfeat + (size_t)b * kstride + (size_t)i0 * c2;
  const _Float16* K1 = kfeat + (size_t)b * kstride + (size_t)i1 * c2;
  const _Float16* K2 = kfeat + (size_t)b * kstride + (size_t)i2 * c2;
  const _Float16* SF = skip + (size_t)b * sstride + (size_t)j * c1;
  _Float16* O = Xo + (size_t)b * xstride + (size_t)j * (c1 + c2);
  for (int c = 0; c < c2; ++c)
    O[c] = (_Float16)(w0 * (float)K0[c] + w1 * (float)K1[c] + w2 * (float)K2[c]);
  for (int c = 0; c < c1; ++c) O[c2 + c] = SF[c];
}

// ---------------- point-0-only final stages ----------------
__device__ __forceinline__ void top3_scan(const float* K, int m, float ux, float uy, float uz,
                                          int* ii, float* w) {
  float d0 = 3.4e38f, d1 = 3.4e38f, d2 = 3.4e38f;
  int i0 = 0, i1 = 0, i2 = 0;
  for (int i = 0; i < m; ++i) {
    float dx = K[i * 3] - ux, dy = K[i * 3 + 1] - uy, dz = K[i * 3 + 2] - uz;
    float d = dx * dx + dy * dy + dz * dz;
    if (d < d0)      { d2 = d1; i2 = i1; d1 = d0; i1 = i0; d0 = d; i0 = i; }
    else if (d < d1) { d2 = d1; i2 = i1; d1 = d; i1 = i; }
    else if (d < d2) { d2 = d; i2 = i; }
  }
  float w0 = 1.f / (d0 + 1e-8f), w1 = 1.f / (d1 + 1e-8f), w2 = 1.f / (d2 + 1e-8f);
  float ws = w0 + w1 + w2;
  ii[0] = i0; ii[1] = i1; ii[2] = i2;
  w[0] = w0 / ws; w[1] = w1 / ws; w[2] = w2 / ws;
}

__global__ void final_near_kernel(const float* __restrict__ pc, const float* __restrict__ cent,
                                  const _Float16* __restrict__ fpfeat,  // col-major [pt][128]
                                  const float* W0, const float* S0, const float* B0,
                                  const float* W1, const float* S1, const float* B1,
                                  const float* W2, const float* S2, const float* B2,
                                  const float* Wfc, const float* Sfc, const float* Bfc,
                                  const float* dir, const float* fdir,
                                  const float* W3h, const float* b3h,
                                  const float* W1h, const float* b1h,
                                  const float* W2h, const float* b2h,
                                  float* __restrict__ out, int m, int npts) {
  int b = blockIdx.x, t = threadIdx.x;  // 128 threads
  __shared__ float xin[132], buf0[128], buf1[128], red[128];
  __shared__ float w[3];
  __shared__ int ii[3];
  const float* U = pc + (size_t)b * npts * 3;
  if (t == 0) top3_scan(cent + (size_t)b * m * 3, m, U[0], U[1], U[2], ii, w);
  __syncthreads();
  {
    const _Float16* KF = fpfeat + (size_t)b * 128 * m;
    xin[t] = w[0] * (float)KF[(size_t)ii[0] * 128 + t] +
             w[1] * (float)KF[(size_t)ii[1] * 128 + t] +
             w[2] * (float)KF[(size_t)ii[2] * 128 + t];
  }
  if (t < 3) xin[128 + t] = U[t];
  __syncthreads();
  { float a = 0; const float* wr = W0 + (size_t)t * 131;
    for (int k = 0; k < 131; ++k) a += wr[k] * xin[k];
    buf0[t] = fmaxf(a * S0[t] + B0[t], 0.f); }
  __syncthreads();
  { float a = 0; const float* wr = W1 + (size_t)t * 128;
    for (int k = 0; k < 128; ++k) a += wr[k] * buf0[k];
    buf1[t] = fmaxf(a * S1[t] + B1[t], 0.f); }
  __syncthreads();
  { float a = 0; const float* wr = W2 + (size_t)t * 128;
    for (int k = 0; k < 128; ++k) a += wr[k] * buf1[k];
    buf0[t] = fmaxf(a * S2[t] + B2[t], 0.f); }
  __syncthreads();
  { float a = 0; const float* wr = Wfc + (size_t)t * 128;
    for (int k = 0; k < 128; ++k) a += wr[k] * buf0[k];
    buf1[t] = fmaxf(a * Sfc[t] + Bfc[t], 0.f); }   // near_feat
  __syncthreads();
  { const float* wr = W3h + (size_t)t * 6; float a = b3h[t];
    for (int k = 0; k < 3; ++k) a += wr[k] * dir[b * 3 + k];
    for (int k = 0; k < 3; ++k) a += wr[3 + k] * fdir[b * 3 + k];
    buf0[t] = a; }                                  // mlp3 out
  __syncthreads();
  { const float* wr = W1h + (size_t)t * 256; float a = b1h[t];
    for (int k = 0; k < 128; ++k) a += wr[k] * buf1[k];
    for (int k = 0; k < 128; ++k) a += wr[128 + k] * buf0[k];
    a = (a > 0.f) ? a : 0.01f * a;                  // leaky relu
    red[t] = W2h[t] * a; }
  __syncthreads();
  for (int s = 64; s > 0; s >>= 1) { if (t < s) red[t] += red[t + s]; __syncthreads(); }
  if (t == 0) out[b] = red[0] + b2h[0];
}

__global__ void final_far_kernel(const float* __restrict__ pc, const float* __restrict__ cent,
                                 const _Float16* __restrict__ fpfeat,  // col-major [pt][128]
                                 const float* W0, const float* S0, const float* B0,
                                 const float* W1, const float* S1, const float* B1,
                                 const float* W2, const float* S2, const float* B2,
                                 const float* Wfc, const float* Sfc, const float* Bfc,
                                 const float* Wh, const float* bh,
                                 float* __restrict__ out, int m, int npts) {
  int b = blockIdx.x, t = threadIdx.x;
  __shared__ float xin[132], buf0[128], buf1[128], red[128];
  __shared__ float w[3];
  __shared__ int ii[3];
  const float* U = pc + (size_t)b * npts * 3;
  if (t == 0) top3_scan(cent + (size_t)b * m * 3, m, U[0], U[1], U[2], ii, w);
  __syncthreads();
  {
    const _Float16* KF = fpfeat + (size_t)b * 128 * m;
    xin[t] = w[0] * (float)KF[(size_t)ii[0] * 128 + t] +
             w[1] * (float)KF[(size_t)ii[1] * 128 + t] +
             w[2] * (float)KF[(size_t)ii[2] * 128 + t];
  }
  if (t < 3) xin[128 + t] = U[t];
  __syncthreads();
  { float a = 0; const float* wr = W0 + (size_t)t * 131;
    for (int k = 0; k < 131; ++k) a += wr[k] * xin[k];
    buf0[t] = fmaxf(a * S0[t] + B0[t], 0.f); }
  __syncthreads();
  { float a = 0; const float* wr = W1 + (size_t)t * 128;
    for (int k = 0; k < 128; ++k) a += wr[k] * buf0[k];
    buf1[t] = fmaxf(a * S1[t] + B1[t], 0.f); }
  __syncthreads();
  { float a = 0; const float* wr = W2 + (size_t)t * 128;
    for (int k = 0; k < 128; ++k) a += wr[k] * buf1[k];
    buf0[t] = fmaxf(a * S2[t] + B2[t], 0.f); }
  __syncthreads();
  { float a = 0; const float* wr = Wfc + (size_t)t * 128;
    for (int k = 0; k < 128; ++k) a += wr[k] * buf0[k];
    buf1[t] = fmaxf(a * Sfc[t] + Bfc[t], 0.f); }   // far_feat
  __syncthreads();
  red[t] = Wh[t] * buf1[t];
  __syncthreads();
  for (int s = 64; s > 0; s >>= 1) { if (t < s) red[t] += red[t + s]; __syncthreads(); }
  if (t == 0) out[b] = red[0] + bh[0];
}

// ---------------- host orchestration ----------------
static const int SA_DIMS0[4] = {6, 32, 32, 64};
static const int SA_DIMS1[4] = {67, 64, 64, 128};
static const int SA_DIMS2[4] = {131, 128, 128, 256};
static const int SA_DIMS3[4] = {259, 256, 256, 512};
static const int FPD_768[3] = {768, 256, 256};
static const int FPD_384[3] = {384, 256, 256};
static const int FPD_320[3] = {320, 256, 128};

extern "C" void kernel_launch(void* const* d_in, const int* in_sizes, int n_in,
                              void* d_out, int out_size, void* d_ws, size_t ws_size,
                              hipStream_t stream) {
  (void)in_sizes; (void)n_in; (void)out_size; (void)ws_size;
  const int B = NBATCH, N = NPTS;
  const float* dir = (const float*)d_in[0];
  const float* fdir = (const float*)d_in[1];
  const float* pc_near = (const float*)d_in[2];
  const float* pc_far = (const float*)d_in[3];
  auto P = [&](int i) { return (const float*)d_in[i]; };

  // params pytree in insertion order: sa_far(4..39) sa_near(40..57) fp_far(58..84)
  // fp_near(85..99) fc_far(100..102) fc_near(103..105) mlp3(106,107) mlp1(108,109)
  // mlp2(110,111) mlp_far(112,113)

  char* base = (char*)d_ws;
  size_t off = 0;
  auto alloc = [&](size_t bytes) -> char* {
    off = (off + 255) & ~(size_t)255;
    char* p = base + off; off += bytes; return p;
  };
  float* mind = (float*)alloc((size_t)B * N * 4);
  int* bidx = (int*)alloc((size_t)B * 1024 * 32 * 4);
  float* x1 = (float*)alloc((size_t)B * 1024 * 3 * 4);
  float* x2 = (float*)alloc((size_t)B * 256 * 3 * 4);
  float* x3 = (float*)alloc((size_t)B * 64 * 3 * 4);
  float* x4 = (float*)alloc((size_t)B * 16 * 3 * 4);
  float* xn1 = (float*)alloc((size_t)B * 1024 * 3 * 4);
  float* xn2 = (float*)alloc((size_t)B * 256 * 3 * 4);
  // all f16 feature maps are column(point)-major: [point][ch]
  _Float16* feat0f = (_Float16*)alloc((size_t)B * N * 3 * 2);
  _Float16* feat0n = (_Float16*)alloc((size_t)B * N * 3 * 2);
  _Float16* f1 = (_Float16*)alloc((size_t)B * 1024 * 64 * 2);
  _Float16* f2 = (_Float16*)alloc((size_t)B * 256 * 128 * 2);
  _Float16* f3 = (_Float16*)alloc((size_t)B * 64 * 256 * 2);
  _Float16* f4 = (_Float16*)alloc((size_t)B * 16 * 512 * 2);
  _Float16* f3p = (_Float16*)alloc((size_t)B * 64 * 256 * 2);
  _Float16* f2p = (_Float16*)alloc((size_t)B * 256 * 256 * 2);
  _Float16* f1p = (_Float16*)alloc((size_t)B * 1024 * 128 * 2);
  _Float16* fn1 = (_Float16*)alloc((size_t)B * 1024 * 64 * 2);
  _Float16* fn2 = (_Float16*)alloc((size_t)B * 256 * 128 * 2);
  _Float16* fn2p = (_Float16*)alloc((size_t)B * 256 * 256 * 2);
  _Float16* fn1p = (_Float16*)alloc((size_t)B * 1024 * 128 * 2);
  const size_t PING = 64 * 32768;  // max intermediate: 32768 cols x 64 ch
  _Float16* pingA = (_Float16*)alloc((size_t)B * PING * 2);
  _Float16* pingB = (_Float16*)alloc((size_t)B * PING * 2);
  _Float16* wbuf = (_Float16*)alloc((size_t)256 * 768 * 2);  // max cout*KPAD (fp768 L0)

  auto run_gemm = [&](int pidx_layer, const _Float16* Xc, _Float16* Yc, int cout, int cin,
                      int kpad, int ncols, size_t xstr, size_t ystr) {
    prep_w_kernel<<<dim3((cout * kpad + 255) / 256), dim3(256), 0, stream>>>(
        P(pidx_layer), wbuf, cout, cin, kpad);
    gemm_wmma_kernel<<<dim3((cout + 63) / 64, ncols / 16, B), dim3(128), 0, stream>>>(
        wbuf, P(pidx_layer + 1), P(pidx_layer + 2), Xc, Yc, cout, kpad, ncols, xstr, ystr);
  };

  auto run_sa = [&](const float* sxyz, int nsrc, const _Float16* feats, int cf,
                    int npoint, float radius, int pidx, const int* dims,
                    float* cent, _Float16* ofeat) {
    fps_kernel<<<dim3(B), dim3(512), 0, stream>>>(sxyz, mind, cent, nsrc, npoint);
    ball_query_kernel<<<dim3((npoint + 127) / 128, B), dim3(128), 0, stream>>>(
        sxyz, cent, bidx, nsrc, npoint, radius * radius, 32);
    int ncols = npoint * 32;
    int kp0 = (dims[0] + 31) & ~31;
    group_kernel<<<dim3((ncols + 255) / 256, B), dim3(256), 0, stream>>>(
        sxyz, cent, bidx, feats, pingA, nsrc, npoint, 32, cf, kp0, (size_t)cf * nsrc, PING);
    _Float16* cur = pingA; _Float16* nxt = pingB;
    int kd[3] = {kp0, dims[1], dims[2]};
    for (int l = 0; l < 3; ++l) {
      run_gemm(pidx + 3 * l, cur, nxt, dims[l + 1], dims[l], kd[l], ncols, PING, PING);
      _Float16* t = cur; cur = nxt; nxt = t;
    }
    int cl = dims[3];
    maxpool_kernel<<<dim3((cl * npoint + 255) / 256, B), dim3(256), 0, stream>>>(
        cur, ofeat, cl, npoint, 32, PING, (size_t)cl * npoint);
  };

  auto run_fp = [&](const float* unk, int n, const float* knw, int m,
                    const _Float16* skip, int c1, const _Float16* kf, int c2,
                    int pidx, const int* dims, int nlayers, _Float16* ofeat) {
    fp_interp_kernel<<<dim3((n + 127) / 128, B), dim3(128), 0, stream>>>(
        unk, knw, skip, kf, pingA, n, m, c1, c2, (size_t)c1 * n, (size_t)c2 * m, PING);
    _Float16* cur = pingA; _Float16* nxt = pingB;
    for (int l = 0; l < nlayers; ++l) {
      bool last = (l == nlayers - 1);
      _Float16* dst = last ? ofeat : nxt;
      size_t ystr = last ? (size_t)dims[nlayers] * n : PING;
      run_gemm(pidx + 3 * l, cur, dst, dims[l + 1], dims[l], dims[l], n, PING, ystr);
      if (!last) { _Float16* t = cur; cur = nxt; nxt = t; }
    }
  };

  // level-0 feature maps (pc duplicated -> feats == xyz channels), point-major [i][3]
  f32_to_f16_kernel<<<dim3((B * N * 3 + 255) / 256), dim3(256), 0, stream>>>(
      pc_far, feat0f, (size_t)B * N * 3);
  f32_to_f16_kernel<<<dim3((B * N * 3 + 255) / 256), dim3(256), 0, stream>>>(
      pc_near, feat0n, (size_t)B * N * 3);

  // FAR branch: SA pyramid
  run_sa(pc_far, N, feat0f, 3, 1024, 0.1f, 4, SA_DIMS0, x1, f1);
  run_sa(x1, 1024, f1, 64, 256, 0.2f, 13, SA_DIMS1, x2, f2);
  run_sa(x2, 256, f2, 128, 64, 0.4f, 22, SA_DIMS2, x3, f3);
  run_sa(x3, 64, f3, 256, 16, 0.8f, 31, SA_DIMS3, x4, f4);
  // FAR FP chain (full down to n=1024; n=10000 level done point-0-only at the end)
  run_fp(x3, 64, x4, 16, f3, 256, f4, 512, 79, FPD_768, 2, f3p);
  run_fp(x2, 256, x3, 64, f2, 128, f3p, 256, 73, FPD_384, 2, f2p);
  run_fp(x1, 1024, x2, 256, f1, 64, f2p, 256, 67, FPD_320, 2, f1p);

  // NEAR branch
  run_sa(pc_near, N, feat0n, 3, 1024, 0.1f, 40, SA_DIMS0, xn1, fn1);
  run_sa(xn1, 1024, fn1, 64, 256, 0.2f, 49, SA_DIMS1, xn2, fn2);
  run_fp(xn2, 256, x3, 64, fn2, 128, f3p, 256, 73, FPD_384, 2, fn2p);  // cross-branch, fp_far[-2]
  run_fp(xn1, 1024, xn2, 256, fn1, 64, fn2p, 256, 94, FPD_320, 2, fn1p);

  // point-0-only final FP + FC + heads
  float* out = (float*)d_out;
  final_near_kernel<<<dim3(B), dim3(128), 0, stream>>>(
      pc_near, xn1, fn1p,
      P(85), P(86), P(87), P(88), P(89), P(90), P(91), P(92), P(93),
      P(103), P(104), P(105), dir, fdir,
      P(106), P(107), P(108), P(109), P(110), P(111), out, 1024, N);
  final_far_kernel<<<dim3(B), dim3(128), 0, stream>>>(
      pc_far, x1, f1p,
      P(58), P(59), P(60), P(61), P(62), P(63), P(64), P(65), P(66),
      P(100), P(101), P(102), P(112), P(113), out + 8, 1024, N);
}
